// MoELayer_73031623901861
// MI455X (gfx1250) — matrix operand order repaired and checked by reading, not compiled, not run on hardware
//
#include <hip/hip_runtime.h>
#include <hip/hip_bf16.h>

// ---------------------------------------------------------------------------
// MoE layer for gfx1250 (MI455X): LN -> router top-2 -> grouped WMMA FFN ->
// weighted combine + residual.
//   * weights pre-transposed/converted to f16 [E][N][K] once per call
//   * double-buffered global->LDS tile pipeline via
//     global_load_async_to_lds_b128 + s_wait_asynccnt (ASYNCcnt, in-order)
//   * matrix math via v_wmma_f32_16x16x32_f16, 4 WMMA / wave / K-step
// ---------------------------------------------------------------------------

#define BDIM  2
#define SDIM  2048
#define HDIM  1024
#define IDIM  4096
#define EDIM  8
#define KSEL  2
#define TDIM  (BDIM * SDIM)        // 4096 tokens
#define PDIM  (TDIM * KSEL)        // 8192 (token, expert) slots

typedef _Float16 v16h __attribute__((ext_vector_type(16)));
typedef _Float16 h8   __attribute__((ext_vector_type(8)));
typedef _Float16 h4   __attribute__((ext_vector_type(4)));
typedef float    v8f  __attribute__((ext_vector_type(8)));

// ---- CDNA5 async global->LDS copy helpers (ASYNCcnt tracked) --------------
__device__ __forceinline__ unsigned lds_off(const void* p) {
  // LDS aperture: byte offset lives in addr[31:0] of the generic pointer.
  return (unsigned)(uintptr_t)p;
}
__device__ __forceinline__ void async_copy_b128(unsigned lds, const void* g) {
  asm volatile("global_load_async_to_lds_b128 %0, %1, off"
               :: "v"(lds), "v"(g) : "memory");
}
// INST_OFFSET is added to BOTH the LDS and the global address (ISA 08 §4.4).
__device__ __forceinline__ void async_copy_b128_o16(unsigned lds, const void* g) {
  asm volatile("global_load_async_to_lds_b128 %0, %1, off offset:16"
               :: "v"(lds), "v"(g) : "memory");
}
__device__ __forceinline__ void async_wait3() {
  asm volatile("s_wait_asynccnt 0x3" ::: "memory");
}
__device__ __forceinline__ void async_wait0() {
  asm volatile("s_wait_asynccnt 0x0" ::: "memory");
}

// ---------------------------------------------------------------------------
// 0) Weight transpose + f32 -> f16 convert:  W[e][k][n] -> Wt[e][n][k]
// ---------------------------------------------------------------------------
__global__ void wtrans_kernel(const float* __restrict__ W,
                              _Float16* __restrict__ Wt,
                              int Ktot, int Ntot) {
  const int e  = blockIdx.z;
  const int n0 = blockIdx.x * 32;
  const int k0 = blockIdx.y * 32;
  const float* src = W + (size_t)e * Ktot * Ntot;
  _Float16*    dst = Wt + (size_t)e * Ktot * Ntot;

  __shared__ float tile[32][33];
  const int x = threadIdx.x, y = threadIdx.y;   // block (32, 8)
#pragma unroll
  for (int j = 0; j < 4; ++j)
    tile[y + j * 8][x] = src[(size_t)(k0 + y + j * 8) * Ntot + n0 + x];
  __syncthreads();
#pragma unroll
  for (int j = 0; j < 4; ++j)
    dst[(size_t)(n0 + y + j * 8) * Ktot + k0 + x] = (_Float16)tile[x][y + j * 8];
}

// ---------------------------------------------------------------------------
// 1) LayerNorm: one block per token, 256 threads, H = 1024.  Writes f16.
// ---------------------------------------------------------------------------
__global__ void ln_kernel(const float* __restrict__ hs,
                          const float* __restrict__ gamma,
                          const float* __restrict__ beta,
                          _Float16* __restrict__ xn) {
  const int t = blockIdx.x;
  const float* x = hs + (size_t)t * HDIM;
  const int i = threadIdx.x * 4;

  float4 xv = *(const float4*)(x + i);
  float s  = xv.x + xv.y + xv.z + xv.w;
  float ss = xv.x * xv.x + xv.y * xv.y + xv.z * xv.z + xv.w * xv.w;

  for (int off = 16; off; off >>= 1) {
    s  += __shfl_xor(s,  off, 32);
    ss += __shfl_xor(ss, off, 32);
  }

  __shared__ float red[16];
  __shared__ float s_mu, s_rs;
  const int wv = threadIdx.x >> 5, lane = threadIdx.x & 31;
  if (lane == 0) { red[wv] = s; red[8 + wv] = ss; }
  __syncthreads();
  if (threadIdx.x == 0) {
    float S = 0.f, SS = 0.f;
    for (int k = 0; k < 8; ++k) { S += red[k]; SS += red[8 + k]; }
    float mu  = S / (float)HDIM;
    float var = SS / (float)HDIM - mu * mu;
    s_mu = mu;
    s_rs = rsqrtf(var + 1e-5f);
  }
  __syncthreads();

  const float mu = s_mu, rs = s_rs;
  float4 gv = *(const float4*)(gamma + i);
  float4 bv = *(const float4*)(beta + i);
  h4 o;
  o[0] = (_Float16)((xv.x - mu) * rs * gv.x + bv.x);
  o[1] = (_Float16)((xv.y - mu) * rs * gv.y + bv.y);
  o[2] = (_Float16)((xv.z - mu) * rs * gv.z + bv.z);
  o[3] = (_Float16)((xv.w - mu) * rs * gv.w + bv.w);
  *(h4*)(xn + (size_t)t * HDIM + i) = o;
}

// ---------------------------------------------------------------------------
// 2) Router: one wave per token.  logits = xn @ Wr + br (E = 8),
//    top-2 (tie -> lower index, matching lax.top_k), softmax over the two.
// ---------------------------------------------------------------------------
__global__ void router_kernel(const _Float16* __restrict__ xn,
                              const float* __restrict__ Wr,
                              const float* __restrict__ br,
                              int* __restrict__ te0, int* __restrict__ te1,
                              float* __restrict__ tw0, float* __restrict__ tw1) {
  const int wv = threadIdx.x >> 5, lane = threadIdx.x & 31;
  const int t = blockIdx.x * 8 + wv;
  const _Float16* x = xn + (size_t)t * HDIM;

  float acc[8];
#pragma unroll
  for (int e = 0; e < 8; ++e) acc[e] = 0.f;

  for (int h = lane; h < HDIM; h += 32) {
    float xv = (float)x[h];
    const float4* wp = (const float4*)(Wr + (size_t)h * EDIM);
    float4 w0 = wp[0], w1 = wp[1];
    acc[0] += xv * w0.x; acc[1] += xv * w0.y;
    acc[2] += xv * w0.z; acc[3] += xv * w0.w;
    acc[4] += xv * w1.x; acc[5] += xv * w1.y;
    acc[6] += xv * w1.z; acc[7] += xv * w1.w;
  }
#pragma unroll
  for (int off = 16; off; off >>= 1)
#pragma unroll
    for (int e = 0; e < 8; ++e) acc[e] += __shfl_xor(acc[e], off, 32);

  if (lane == 0) {
#pragma unroll
    for (int e = 0; e < 8; ++e) acc[e] += br[e];
    int i0 = 0; float v0 = acc[0];
#pragma unroll
    for (int e = 1; e < 8; ++e) if (acc[e] > v0) { v0 = acc[e]; i0 = e; }
    int i1 = -1; float v1 = -3.0e38f;
#pragma unroll
    for (int e = 0; e < 8; ++e) if (e != i0 && acc[e] > v1) { v1 = acc[e]; i1 = e; }
    float e1 = __expf(v1 - v0);          // v0 >= v1
    float w0 = 1.f / (1.f + e1);
    float w1 = e1 / (1.f + e1);
    te0[t] = i0; te1[t] = i1; tw0[t] = w0; tw1[t] = w1;
  }
}

// ---------------------------------------------------------------------------
// 3) Per-expert token lists (deterministic, no atomics).
// ---------------------------------------------------------------------------
__global__ void count_kernel(const int* __restrict__ te0, const int* __restrict__ te1,
                             int* __restrict__ cnt) {
  const int e = blockIdx.x;
  int c = 0;
  for (int t = threadIdx.x; t < TDIM; t += blockDim.x)
    c += (te0[t] == e) + (te1[t] == e);
  for (int off = 16; off; off >>= 1) c += __shfl_xor(c, off, 32);
  __shared__ int red[8];
  const int wv = threadIdx.x >> 5, lane = threadIdx.x & 31;
  if (lane == 0) red[wv] = c;
  __syncthreads();
  if (threadIdx.x == 0) {
    int tot = 0;
    for (int k = 0; k < 8; ++k) tot += red[k];
    cnt[e] = tot;
  }
}

__global__ void offsets_kernel(const int* __restrict__ cnt, int* __restrict__ off) {
  if (threadIdx.x == 0) {
    int o = 0;
    for (int e = 0; e < EDIM; ++e) { off[e] = o; o += cnt[e]; }
    off[EDIM] = o;
  }
}

__global__ void build_kernel(const int* __restrict__ te0, const int* __restrict__ te1,
                             const int* __restrict__ off,
                             int* __restrict__ rowlist,
                             int* __restrict__ posA, int* __restrict__ posB) {
  const int e = blockIdx.x;
  __shared__ int wsum[8];
  __shared__ int running;
  if (threadIdx.x == 0) running = off[e];
  __syncthreads();

  for (int base = 0; base < TDIM; base += 256) {
    const int t = base + threadIdx.x;
    const bool m0 = (te0[t] == e);
    const bool m1 = (te1[t] == e);
    const bool match = m0 || m1;               // top-2 indices distinct
    const unsigned mask = (unsigned)__ballot(match);
    const int lane = threadIdx.x & 31, wv = threadIdx.x >> 5;
    if (lane == 0) wsum[wv] = __popc(mask);
    __syncthreads();
    int woff = 0;
    for (int k = 0; k < wv; ++k) woff += wsum[k];
    const int pre = __popc(mask & ((1u << lane) - 1u));
    if (match) {
      const int pos = running + woff + pre;
      rowlist[pos] = t;
      if (m0) posA[t] = pos; else posB[t] = pos;
    }
    __syncthreads();
    if (threadIdx.x == 0) {
      int tot = 0;
      for (int k = 0; k < 8; ++k) tot += wsum[k];
      running += tot;
    }
    __syncthreads();
  }
}

// ---------------------------------------------------------------------------
// 4) Grouped WMMA GEMM.  Block tile 128x64, K-step 32, 256 threads = 8 waves.
//    Wave (wr = wid&3, wc = wid>>2) owns a 32x32 patch: 2 A-frags x 2 B-frags
//    -> 4 v_wmma per K-step.  Double-buffered LDS tiles filled with
//    global_load_async_to_lds_b128; tile k+1 streams in while tile k computes.
//    Every wave issues exactly 3 async copies per tile (A rows clamped, never
//    skipped), so s_wait_asynccnt<=3 guarantees the older tile has landed
//    (async loads complete in order, ISA 08 §4.1).
// ---------------------------------------------------------------------------
#define BM 128
#define BN 64
#define BK 32
#define LDSK 40   // BK + 8 halves padding: 80-byte row stride, 16B aligned
#define ABYTES (BM * LDSK * 2)   // 10240 B per A buffer
#define BBYTES (BN * LDSK * 2)   //  5120 B per B buffer

template <bool GATHER, bool GELU, int KTOT, int NTOT, bool F16OUT>
__global__ __launch_bounds__(256)
void gemm_kernel(const _Float16* __restrict__ A,
                 const _Float16* __restrict__ Wt,    // [E][NTOT][KTOT] f16
                 const float* __restrict__ bias,
                 const int* __restrict__ rowlist,
                 const int* __restrict__ off,
                 const int* __restrict__ cnt,
                 _Float16* __restrict__ outH,
                 float* __restrict__ outF) {
  const int e = blockIdx.z;
  const int c = cnt[e];
  const int m0 = blockIdx.y * BM;
  if (m0 >= c) return;
  const int o  = off[e];
  const int n0 = blockIdx.x * BN;
  const _Float16* WtE = Wt + (size_t)e * KTOT * NTOT;

  __shared__ __align__(16) _Float16 As[2][BM][LDSK];
  __shared__ __align__(16) _Float16 Bs[2][BN][LDSK];

  const int tid  = threadIdx.x;
  const int lane = tid & 31;
  const int wid  = tid >> 5;
  const int wr   = wid & 3;        // 0..3 row strip (32 rows)
  const int wc   = wid >> 2;       // 0..1 col strip (32 cols)
  const int hi   = lane >> 4;      // 0/1
  const int lrow = lane & 15;

  // ---- A-tile copy: 128 rows x 32 halves; 32 B / thread, row CLAMPED so
  //      every wave always issues the same copy count (garbage rows are
  //      independent and their outputs are never stored).
  const int ar = tid >> 1;               // 0..127
  const int ak = (tid & 1) * 16;         // 0 or 16 halves
  const int mrow = (m0 + ar < c) ? (m0 + ar) : (c - 1);
  const int arow = GATHER ? rowlist[o + mrow] : (o + mrow);
  const _Float16* aG = A + (size_t)arow * KTOT + ak;
  const unsigned  aL = lds_off(&As[0][ar][ak]);

  // ---- B-tile copy: 64 rows x 32 halves; 16 B / thread
  const int brow = tid >> 2;             // 0..63
  const int bk   = (tid & 3) * 8;        // 0/8/16/24
  const _Float16* bG = WtE + (size_t)(n0 + brow) * KTOT + bk;
  const unsigned  bL = lds_off(&Bs[0][brow][bk]);

  v8f acc[2][2];
#pragma unroll
  for (int i = 0; i < 2; ++i)
#pragma unroll
    for (int j = 0; j < 2; ++j)
      acc[i][j] = (v8f){0.f, 0.f, 0.f, 0.f, 0.f, 0.f, 0.f, 0.f};

  // issue one tile's 3 async copies into buffer `buf`
  auto issue = [&](int kt, int buf) {
    const _Float16* ag = aG + kt * BK;
    const unsigned abuf = aL + buf * ABYTES;
    async_copy_b128(abuf, ag);
    async_copy_b128_o16(abuf, ag);           // +16 B on LDS and global
    async_copy_b128(bL + buf * BBYTES, bG + kt * BK);
  };

  // compute one K-step from buffer `buf` (buf is a compile-time constant so
  // the double-buffer offset folds into the ds_load immediate offsets)
  auto compute = [&](auto bufc) {
    constexpr int buf = decltype(bufc)::value;
    h8 t0, t1;
    t0 = *(const h8*)&As[buf][wr * 32 + lrow][hi * 8];
    t1 = *(const h8*)&As[buf][wr * 32 + lrow][16 + hi * 8];
    v16h af0 = __builtin_shufflevector(t0, t1, 0, 1, 2, 3, 4, 5, 6, 7,
                                       8, 9, 10, 11, 12, 13, 14, 15);
    t0 = *(const h8*)&As[buf][wr * 32 + 16 + lrow][hi * 8];
    t1 = *(const h8*)&As[buf][wr * 32 + 16 + lrow][16 + hi * 8];
    v16h af1 = __builtin_shufflevector(t0, t1, 0, 1, 2, 3, 4, 5, 6, 7,
                                       8, 9, 10, 11, 12, 13, 14, 15);
    t0 = *(const h8*)&Bs[buf][wc * 32 + lrow][hi * 8];
    t1 = *(const h8*)&Bs[buf][wc * 32 + lrow][16 + hi * 8];
    v16h bf0 = __builtin_shufflevector(t0, t1, 0, 1, 2, 3, 4, 5, 6, 7,
                                       8, 9, 10, 11, 12, 13, 14, 15);
    t0 = *(const h8*)&Bs[buf][wc * 32 + 16 + lrow][hi * 8];
    t1 = *(const h8*)&Bs[buf][wc * 32 + 16 + lrow][16 + hi * 8];
    v16h bf1 = __builtin_shufflevector(t0, t1, 0, 1, 2, 3, 4, 5, 6, 7,
                                       8, 9, 10, 11, 12, 13, 14, 15);

    acc[0][0] = __builtin_amdgcn_wmma_f32_16x16x32_f16(false, af0, false, bf0,
                                                       (short)0, acc[0][0], false, false);
    acc[0][1] = __builtin_amdgcn_wmma_f32_16x16x32_f16(false, af0, false, bf1,
                                                       (short)0, acc[0][1], false, false);
    acc[1][0] = __builtin_amdgcn_wmma_f32_16x16x32_f16(false, af1, false, bf0,
                                                       (short)0, acc[1][0], false, false);
    acc[1][1] = __builtin_amdgcn_wmma_f32_16x16x32_f16(false, af1, false, bf1,
                                                       (short)0, acc[1][1], false, false);
  };

  constexpr int NT = KTOT / BK;          // 32 (GEMM1) or 128 (GEMM2), even
  issue(0, 0);                           // prologue: tile 0 -> buf 0

  for (int kt = 0; kt < NT; kt += 2) {
    // ---- phase 0: stream tile kt+1 into buf1, compute tile kt from buf0
    issue(kt + 1, 1);                    // kt+1 <= NT-1 always valid
    async_wait3();                       // older 3 copies (buf0) have landed
    __syncthreads();
    compute(std::integral_constant<int, 0>{});
    __syncthreads();

    // ---- phase 1: stream tile kt+2 into buf0, compute tile kt+1 from buf1
    if (kt + 2 < NT) {
      issue(kt + 2, 0);
      if (kt + 4 < NT)
        __builtin_prefetch(bG + (kt + 4) * BK, 0, 3);
      async_wait3();
    } else {
      async_wait0();
    }
    __syncthreads();
    compute(std::integral_constant<int, 1>{});
    __syncthreads();
  }

  // ---- epilogue: bias (+GELU), store
#pragma unroll
  for (int j = 0; j < 2; ++j) {
    const int ncol = n0 + wc * 32 + j * 16 + lrow;
    const float bv = bias[(size_t)e * NTOT + ncol];
#pragma unroll
    for (int i = 0; i < 2; ++i) {
#pragma unroll
      for (int r = 0; r < 8; ++r) {
        const int m = m0 + wr * 32 + i * 16 + r + 8 * hi;
        if (m < c) {
          float v = acc[i][j][r] + bv;
          if (GELU)
            v = 0.5f * v * (1.f + erff(v * 0.70710678118654752f));
          const size_t base = (size_t)(o + m) * NTOT;
          if (F16OUT) outH[base + ncol] = (_Float16)v;
          else        outF[base + ncol] = v;
        }
      }
    }
  }
}

// ---------------------------------------------------------------------------
// 5) Combine: out = hidden + w0 * y[posA] + w1 * y[posB]
// ---------------------------------------------------------------------------
__global__ void combine_kernel(const float* __restrict__ hs,
                               const float* __restrict__ tw0,
                               const float* __restrict__ tw1,
                               const int* __restrict__ posA,
                               const int* __restrict__ posB,
                               const float* __restrict__ y,
                               float* __restrict__ out) {
  const int t = blockIdx.x;
  const int i = threadIdx.x * 4;
  const float w0 = tw0[t], w1 = tw1[t];
  const float* ya = y + (size_t)posA[t] * HDIM + i;
  const float* yb = y + (size_t)posB[t] * HDIM + i;
  float4 hv = *(const float4*)(hs + (size_t)t * HDIM + i);
  float4 av = *(const float4*)ya;
  float4 bv = *(const float4*)yb;
  float4 ov;
  ov.x = hv.x + w0 * av.x + w1 * bv.x;
  ov.y = hv.y + w0 * av.y + w1 * bv.y;
  ov.z = hv.z + w0 * av.z + w1 * bv.z;
  ov.w = hv.w + w0 * av.w + w1 * bv.w;
  *(float4*)(out + (size_t)t * HDIM + i) = ov;
}

// ---------------------------------------------------------------------------
// Launcher
// ---------------------------------------------------------------------------
extern "C" void kernel_launch(void* const* d_in, const int* in_sizes, int n_in,
                              void* d_out, int out_size, void* d_ws, size_t ws_size,
                              hipStream_t stream) {
  const float* hs    = (const float*)d_in[0];
  const float* gamma = (const float*)d_in[1];
  const float* beta  = (const float*)d_in[2];
  const float* Wr    = (const float*)d_in[3];
  const float* br    = (const float*)d_in[4];
  const float* W1    = (const float*)d_in[5];
  const float* b1    = (const float*)d_in[6];
  const float* W2    = (const float*)d_in[7];
  const float* b2    = (const float*)d_in[8];
  float* out = (float*)d_out;

  // workspace carve-up (~232 MB total)
  char* ws = (char*)d_ws;
  size_t cur = 0;
  auto carve = [&](size_t bytes) {
    size_t r = cur;
    cur = (cur + bytes + 255) & ~(size_t)255;
    return r;
  };
  _Float16* xn   = (_Float16*)(ws + carve((size_t)TDIM * HDIM * 2));
  _Float16* hbuf = (_Float16*)(ws + carve((size_t)PDIM * IDIM * 2));
  float*    ybuf = (float*)   (ws + carve((size_t)PDIM * HDIM * 4));
  _Float16* wt1  = (_Float16*)(ws + carve((size_t)EDIM * HDIM * IDIM * 2));
  _Float16* wt2  = (_Float16*)(ws + carve((size_t)EDIM * IDIM * HDIM * 2));
  int*   te0     = (int*)  (ws + carve(TDIM * 4));
  int*   te1     = (int*)  (ws + carve(TDIM * 4));
  float* tw0     = (float*)(ws + carve(TDIM * 4));
  float* tw1     = (float*)(ws + carve(TDIM * 4));
  int*   cnt     = (int*)  (ws + carve(EDIM * 4));
  int*   off     = (int*)  (ws + carve((EDIM + 1) * 4));
  int*   rowlist = (int*)  (ws + carve(PDIM * 4));
  int*   posA    = (int*)  (ws + carve(TDIM * 4));
  int*   posB    = (int*)  (ws + carve(TDIM * 4));
  (void)ws_size; (void)in_sizes; (void)n_in; (void)out_size;

  // weight transpose/convert: W1[e][H][I] -> wt1[e][I][H]; W2[e][I][H] -> wt2[e][H][I]
  wtrans_kernel<<<dim3(IDIM / 32, HDIM / 32, EDIM), dim3(32, 8), 0, stream>>>(
      W1, wt1, HDIM, IDIM);
  wtrans_kernel<<<dim3(HDIM / 32, IDIM / 32, EDIM), dim3(32, 8), 0, stream>>>(
      W2, wt2, IDIM, HDIM);

  ln_kernel<<<TDIM, 256, 0, stream>>>(hs, gamma, beta, xn);
  router_kernel<<<TDIM / 8, 256, 0, stream>>>(xn, Wr, br, te0, te1, tw0, tw1);
  count_kernel<<<EDIM, 256, 0, stream>>>(te0, te1, cnt);
  offsets_kernel<<<1, 32, 0, stream>>>(cnt, off);
  build_kernel<<<EDIM, 256, 0, stream>>>(te0, te1, off, rowlist, posA, posB);

  // GEMM1: h = gelu(x_gathered @ W1[e] + b1[e])   [P, I] f16
  gemm_kernel<true, true, HDIM, IDIM, true>
      <<<dim3(IDIM / BN, TDIM / BM, EDIM), 256, 0, stream>>>(
          xn, wt1, b1, rowlist, off, cnt, hbuf, nullptr);

  // GEMM2: y = h @ W2[e] + b2[e]                  [P, H] f32
  gemm_kernel<false, false, IDIM, HDIM, false>
      <<<dim3(HDIM / BN, TDIM / BM, EDIM), 256, 0, stream>>>(
          hbuf, wt2, b2, nullptr, off, cnt, nullptr, ybuf);

  combine_kernel<<<TDIM, 256, 0, stream>>>(hs, tw0, tw1, posA, posB, ybuf, out);
}
